// Model_44074954392060
// MI455X (gfx1250) — compile-verified, matrix operand order checked by reading, not run on hardware
//
#include <hip/hip_runtime.h>

#define NN 25000
#define EE 400000
#define RRR 100
#define DDIM 256
#define HHEAD 8
#define DHH 32
#define LLAY 3
#define BBATCH 4096
#define NUMDRUGS 2000
#define ALPHA_C 0.05f
#define HOPS_C 2
#define TOPK_C 10
#define NEGV -1e9f

typedef __attribute__((ext_vector_type(16))) __bf16 v16bf;
typedef __attribute__((ext_vector_type(2)))  __bf16 bf16x2;
typedef __attribute__((ext_vector_type(8)))  float  v8f;

union FragU { unsigned u[8]; v16bf v; };

// ---------- order-preserving float<->uint encoding for atomic max on f32 ----------
__device__ __forceinline__ unsigned f2ord(float f) {
    unsigned u = __float_as_uint(f);
    return (u & 0x80000000u) ? ~u : (u | 0x80000000u);
}
__device__ __forceinline__ float ord2f(unsigned u) {
    return __uint_as_float((u & 0x80000000u) ? (u & 0x7FFFFFFFu) : ~u);
}

// ---------- LDS-staged bf16 WMMA GEMM: C[M,Nc] = act(A[M,K] @ B[K,Nc] + bias) ----------
// block tile 128x64, 8 waves in 4x2; each wave computes a 32x32 macro-tile (4 WMMAs/K-step).
// K multiple of 32, Nc multiple of 64. M edge handled by clamping A rows + masking stores.
#define BM 128
#define BN 64
#define BK 32
#define LDP 17   // padded LDS row stride (dwords) to avoid bank conflicts

__global__ __launch_bounds__(256) void gemm_wmma_kernel(
        const float* __restrict__ A, const float* __restrict__ B,
        const float* __restrict__ bias, float* __restrict__ C,
        int M, int K, int Nc, int act) {
    __shared__ unsigned As[BM * LDP];   // [row][k2]: packed bf16 (k,k+1)
    __shared__ unsigned Bs[BN * LDP];   // [col][k2]: packed bf16 (k,k+1)

    const int t = threadIdx.x;
    const int wave = t >> 5, lane = t & 31;
    const int wm = wave >> 1, wn = wave & 1;      // 4x2 wave grid
    const int half = lane >> 4, idx = lane & 15;
    const int blockRow = blockIdx.x * BM;
    const int blockCol = blockIdx.y * BN;

    v8f acc[2][2] = {};

    for (int k0 = 0; k0 < K; k0 += BK) {
        // ---- stage A: 128x32 fp32 -> packed bf16 in LDS (4 float4 per thread) ----
#pragma unroll
        for (int i = 0; i < 4; ++i) {
            int q = t + i * 256;            // 0..1023 float4 slots
            int r = q >> 3;                 // 8 float4 per 32-wide row
            int cc = (q & 7) << 2;          // col within tile
            int gr = blockRow + r; if (gr >= M) gr = M - 1;   // clamp (masked at store)
            const float4 f = *(const float4*)(A + (size_t)gr * K + k0 + cc);
            bf16x2 p0; p0.x = (__bf16)f.x; p0.y = (__bf16)f.y;
            bf16x2 p1; p1.x = (__bf16)f.z; p1.y = (__bf16)f.w;
            As[r * LDP + (cc >> 1)]     = __builtin_bit_cast(unsigned, p0);
            As[r * LDP + (cc >> 1) + 1] = __builtin_bit_cast(unsigned, p1);
        }
        // ---- stage B transposed: Bs[n][k2] = pack(B[k0+2k2][n], B[k0+2k2+1][n]) ----
#pragma unroll
        for (int i = 0; i < 4; ++i) {
            int p = t + i * 256;            // 0..1023 (64 n x 16 k2)
            int n = p & 63;
            int k2 = p >> 6;                // 0..15
            const float* bp = B + (size_t)(k0 + 2 * k2) * Nc + blockCol + n;
            bf16x2 pb; pb.x = (__bf16)bp[0]; pb.y = (__bf16)bp[Nc];
            Bs[n * LDP + k2] = __builtin_bit_cast(unsigned, pb);
        }
        __syncthreads();

        // ---- fragments from LDS ----
        FragU fa[2], fb[2];
#pragma unroll
        for (int tm = 0; tm < 2; ++tm) {
            int row = wm * 32 + tm * 16 + idx;
#pragma unroll
            for (int j = 0; j < 4; ++j) {
                fa[tm].u[j]     = As[row * LDP + j + 4 * half];       // K 0..15 region
                fa[tm].u[4 + j] = As[row * LDP + 8 + j + 4 * half];   // K 16..31 region
            }
        }
#pragma unroll
        for (int tn = 0; tn < 2; ++tn) {
            int col = wn * 32 + tn * 16 + idx;
#pragma unroll
            for (int j = 0; j < 4; ++j) {
                fb[tn].u[j]     = Bs[col * LDP + j + 4 * half];
                fb[tn].u[4 + j] = Bs[col * LDP + 8 + j + 4 * half];
            }
        }
#pragma unroll
        for (int tm = 0; tm < 2; ++tm)
#pragma unroll
            for (int tn = 0; tn < 2; ++tn)
                acc[tm][tn] = __builtin_amdgcn_wmma_f32_16x16x32_bf16(
                    false, fa[tm].v, false, fb[tn].v, (short)0, acc[tm][tn], false, false);
        __syncthreads();
    }

    // ---- epilogue: bias + activation + masked store ----
#pragma unroll
    for (int tn = 0; tn < 2; ++tn) {
        int col = blockCol + wn * 32 + tn * 16 + idx;
        float bv = bias ? bias[col] : 0.f;
#pragma unroll
        for (int tm = 0; tm < 2; ++tm) {
#pragma unroll
            for (int j = 0; j < 8; ++j) {
                int row = blockRow + wm * 32 + tm * 16 + half * 8 + j;
                if (row < M) {
                    float v = acc[tm][tn][j] + bv;
                    if (act == 1) v = fmaxf(v, 0.f);
                    else if (act == 2) v = 1.f / (1.f + __expf(-v));
                    C[(size_t)row * Nc + col] = v;
                }
            }
        }
    }
}

// ---------- fill ----------
__global__ void fill_u32_kernel(unsigned* __restrict__ p, unsigned v, int n) {
    int t = blockIdx.x * blockDim.x + threadIdx.x;
    if (t < n) p[t] = v;
}

// ---------- per-(node,head) attention projections: s = <x[n,h,:], a[h,:]> ----------
__global__ void proj_scores_kernel(const float* __restrict__ X,
                                   const float* __restrict__ a1, const float* __restrict__ a2,
                                   float* __restrict__ o1, float* __restrict__ o2, int rows) {
    int t = blockIdx.x * blockDim.x + threadIdx.x;
    if (t >= rows * HHEAD) return;
    int n = t >> 3, hh = t & 7;
    const float* x = X + (size_t)n * DDIM + hh * DHH;
    const float* v1 = a1 + hh * DHH;
    float s1 = 0.f, s2 = 0.f;
    if (a2) {
        const float* v2 = a2 + hh * DHH;
        for (int i = 0; i < DHH; ++i) { float xv = x[i]; s1 += xv * v1[i]; s2 += xv * v2[i]; }
        o1[t] = s1; o2[t] = s2;
    } else {
        for (int i = 0; i < DHH; ++i) s1 += x[i] * v1[i];
        o1[t] = s1;
    }
}

// ---------- edge logits: e = leaky_relu(s_src[src]+s_dst[dst]+s_rel[rel], 0.2) ----------
__global__ void edge_logits_kernel(const float* __restrict__ ssrc, const float* __restrict__ sdst,
                                   const float* __restrict__ srel,
                                   const int* __restrict__ src, const int* __restrict__ dst,
                                   const int* __restrict__ rt,
                                   float* __restrict__ e, float* __restrict__ sw) {
    int t = blockIdx.x * blockDim.x + threadIdx.x;
    if (t >= EE * HHEAD) return;
    int ed = t >> 3, h = t & 7;
    float v = ssrc[src[ed] * HHEAD + h] + sdst[dst[ed] * HHEAD + h] + srel[rt[ed] * HHEAD + h];
    v = v > 0.f ? v : 0.2f * v;
    e[t] = v; sw[t] = v;
}

// ---------- segment max over edges into per-(dst,head) slot ----------
__global__ void seg_max_kernel(const float* __restrict__ s, const int* __restrict__ dst,
                               unsigned* __restrict__ bits) {
    int t = blockIdx.x * blockDim.x + threadIdx.x;
    if (t >= EE * HHEAD) return;
    int ed = t >> 3, h = t & 7;
    atomicMax(&bits[dst[ed] * HHEAD + h], f2ord(s[t]));
}

// ---------- suppress current max values (for iterative kth-max) ----------
__global__ void suppress_kernel(float* __restrict__ s, const int* __restrict__ dst,
                                const unsigned* __restrict__ bits) {
    int t = blockIdx.x * blockDim.x + threadIdx.x;
    if (t >= EE * HHEAD) return;
    int ed = t >> 3, h = t & 7;
    float thr = ord2f(bits[dst[ed] * HHEAD + h]);
    if (s[t] >= thr) s[t] = NEGV;
}

// ---------- masked segment max for softmax numerator ----------
__global__ void seg_max_keep_kernel(const float* __restrict__ e, const int* __restrict__ dst,
                                    const unsigned* __restrict__ thr, unsigned* __restrict__ mb) {
    int t = blockIdx.x * blockDim.x + threadIdx.x;
    if (t >= EE * HHEAD) return;
    int ed = t >> 3, h = t & 7;
    int slot = dst[ed] * HHEAD + h;
    float ev = e[t];
    if (ev >= ord2f(thr[slot])) atomicMax(&mb[slot], f2ord(ev));
}

// ---------- exp + denom accumulate ----------
__global__ void exp_pass_kernel(const float* __restrict__ e, const int* __restrict__ dst,
                                const unsigned* __restrict__ thr, const unsigned* __restrict__ mb,
                                float* __restrict__ attn, float* __restrict__ denom) {
    int t = blockIdx.x * blockDim.x + threadIdx.x;
    if (t >= EE * HHEAD) return;
    int ed = t >> 3, h = t & 7;
    int slot = dst[ed] * HHEAD + h;
    float ev = e[t];
    float ex = 0.f;
    if (ev >= ord2f(thr[slot])) {
        ex = __expf(ev - ord2f(mb[slot]));
        atomicAdd(&denom[slot], ex);
    }
    attn[t] = ex;
}

__global__ void attn_norm_kernel(float* __restrict__ attn, const int* __restrict__ dst,
                                 const float* __restrict__ denom) {
    int t = blockIdx.x * blockDim.x + threadIdx.x;
    if (t >= EE * HHEAD) return;
    int ed = t >> 3, h = t & 7;
    attn[t] = attn[t] / (denom[dst[ed] * HHEAD + h] + 1e-9f);
}

// ---------- diffusion scatter: agg[dst] += attn * feat[src]; 64 threads/edge, float4 ----------
__global__ void diffuse_scatter_kernel(const float* __restrict__ feat, const float* __restrict__ attn,
                                       const int* __restrict__ src, const int* __restrict__ dst,
                                       float* __restrict__ agg) {
    int gid = blockIdx.x * blockDim.x + threadIdx.x;
    if (gid >= EE * 64) return;
    int e = gid >> 6;
    int c = gid & 63;                 // chunk of 4 floats
    int head = c >> 3;                // (c*4)/32
    float a = attn[(size_t)e * HHEAD + head];
    if (a == 0.f) return;             // pruned edge: skip gather + atomics
    int s = src[e], d = dst[e];
    const float4 f = *(const float4*)(feat + (size_t)s * DDIM + c * 4);
    float* o = agg + (size_t)d * DDIM + c * 4;
    atomicAdd(o + 0, a * f.x); atomicAdd(o + 1, a * f.y);
    atomicAdd(o + 2, a * f.z); atomicAdd(o + 3, a * f.w);
}

// ---------- feat = ca*agg + cb*feat0 ----------
__global__ void axpby_kernel(const float* __restrict__ a, const float* __restrict__ b,
                             float* __restrict__ o, float ca, float cb, int n) {
    int t = blockIdx.x * blockDim.x + threadIdx.x;
    if (t < n) o[t] = ca * a[t] + cb * b[t];
}

// ---------- h = LayerNorm(h + add) row-wise, one block(256) per row ----------
__global__ void residual_ln_kernel(float* __restrict__ h, const float* __restrict__ add,
                                   const float* __restrict__ g, const float* __restrict__ b) {
    __shared__ float rsum[DDIM], rsq[DDIM];
    int row = blockIdx.x, c = threadIdx.x;
    size_t i = (size_t)row * DDIM + c;
    float x = h[i] + add[i];
    rsum[c] = x; rsq[c] = x * x;
    __syncthreads();
    for (int s2 = DDIM / 2; s2 > 0; s2 >>= 1) {
        if (c < s2) { rsum[c] += rsum[c + s2]; rsq[c] += rsq[c + s2]; }
        __syncthreads();
    }
    float mu = rsum[0] * (1.f / DDIM);
    float var = rsq[0] * (1.f / DDIM) - mu * mu;
    h[i] = (x - mu) * rsqrtf(var + 1e-5f) * g[c] + b[c];
}

// ---------- X0[b] = h[drug_b] + h[NUMDRUGS + prot_b] ----------
__global__ void gather_pairs_kernel(const float* __restrict__ h, const int* __restrict__ xd,
                                    const int* __restrict__ xp, float* __restrict__ X0) {
    int t = blockIdx.x * blockDim.x + threadIdx.x;
    if (t >= BBATCH * DDIM) return;
    int bI = t >> 8, c = t & 255;
    X0[t] = h[(size_t)xd[bI] * DDIM + c] + h[(size_t)(NUMDRUGS + xp[bI]) * DDIM + c];
}

// ---------- final: out[b] = sigmoid(<z2[b], w> + bias); one wave per row ----------
__global__ void final_layer_kernel(const float* __restrict__ z2, const float* __restrict__ w,
                                   const float* __restrict__ b, float* __restrict__ out) {
    int gwarp = (blockIdx.x * blockDim.x + threadIdx.x) >> 5;
    int lane = threadIdx.x & 31;
    if (gwarp >= BBATCH) return;
    const float* zr = z2 + (size_t)gwarp * 512;
    float s = 0.f;
    for (int i = lane; i < 512; i += 32) s += zr[i] * w[i];
    for (int off = 16; off > 0; off >>= 1) s += __shfl_down(s, off, 32);
    if (lane == 0) out[gwarp] = 1.f / (1.f + __expf(-(s + b[0])));
}

// =============================================================================
extern "C" void kernel_launch(void* const* d_in, const int* in_sizes, int n_in,
                              void* d_out, int out_size, void* d_ws, size_t ws_size,
                              hipStream_t stream) {
    const float* ent    = (const float*)d_in[0];
    const float* relemb = (const float*)d_in[1];
    const float* W_ent  = (const float*)d_in[2];
    const float* W_rel  = (const float*)d_in[3];
    const float* Wq     = (const float*)d_in[4];
    const float* Wr     = (const float*)d_in[5];
    const float* a_src  = (const float*)d_in[6];
    const float* a_dst  = (const float*)d_in[7];
    const float* a_rel  = (const float*)d_in[8];
    const float* ffn_w1 = (const float*)d_in[9];
    const float* ffn_b1 = (const float*)d_in[10];
    const float* ffn_w2 = (const float*)d_in[11];
    const float* ffn_b2 = (const float*)d_in[12];
    const float* ln1_g  = (const float*)d_in[13];
    const float* ln1_b  = (const float*)d_in[14];
    const float* ln2_g  = (const float*)d_in[15];
    const float* ln2_b  = (const float*)d_in[16];
    const float* cls_w1 = (const float*)d_in[17];
    const float* cls_b1 = (const float*)d_in[18];
    const float* cls_w2 = (const float*)d_in[19];
    const float* cls_b2 = (const float*)d_in[20];
    const float* cls_w3 = (const float*)d_in[21];
    const float* cls_b3 = (const float*)d_in[22];
    const int* src = (const int*)d_in[23];
    const int* dst = (const int*)d_in[24];
    const int* rt  = (const int*)d_in[25];
    const int* xd  = (const int*)d_in[26];
    const int* xp  = (const int*)d_in[27];
    float* out = (float*)d_out;

    // ---- workspace layout (floats) ----
    float* ws = (float*)d_ws;
    size_t o = 0;
    float* h    = ws + o; o += (size_t)NN * DDIM;     // node features
    float* hp   = ws + o; o += (size_t)NN * DDIM;     // h @ Wq (= feat0); reused as z1
    float* feat = ws + o; o += (size_t)NN * DDIM;     // diffusion state; reused as ffn tmp / X0
    float* agg  = ws + o; o += (size_t)NN * DDIM;     // hop accumulator; reused as ff / z2
    float* rm   = ws + o; o += (size_t)RRR * DDIM;    // relation embedding mapped
    float* rp   = ws + o; o += (size_t)RRR * DDIM;    // r @ Wr per layer
    float* ssrc = ws + o; o += (size_t)NN * HHEAD;
    float* sdst = ws + o; o += (size_t)NN * HHEAD;
    float* srel = ws + o; o += (size_t)RRR * HHEAD;
    float* ebuf = ws + o; o += (size_t)EE * HHEAD;    // edge logits
    float* sw   = ws + o; o += (size_t)EE * HHEAD;    // working copy for kth-max
    float* attn = ws + o; o += (size_t)EE * HHEAD;    // ex -> normalized attention
    unsigned* thr   = (unsigned*)(ws + o); o += (size_t)NN * HHEAD;
    unsigned* mbits = (unsigned*)(ws + o); o += (size_t)NN * HHEAD;
    float*    denom = ws + o; o += (size_t)NN * HHEAD;

    const int TB = 256;
    const int nEH = EE * HHEAD;
    const int gEH = (nEH + TB - 1) / TB;
    const int nNH = NN * HHEAD;
    const int gNH = (nNH + TB - 1) / TB;
    const unsigned ENC_NEG = ~__builtin_bit_cast(unsigned, NEGV);  // f2ord(-1e9)

    auto gemm = [&](const float* A, const float* Bm, const float* bias, float* C,
                    int M, int K, int Nc, int act) {
        dim3 grid((M + BM - 1) / BM, Nc / BN);
        gemm_wmma_kernel<<<grid, dim3(256), 0, stream>>>(A, Bm, bias, C, M, K, Nc, act);
    };

    // ---- input maps ----
    gemm(ent, W_ent, nullptr, h, NN, DDIM, DDIM, 0);
    gemm(relemb, W_rel, nullptr, rm, RRR, DDIM, DDIM, 0);

    for (int l = 0; l < LLAY; ++l) {
        const float* Wq_l = Wq + (size_t)l * DDIM * DDIM;
        const float* Wr_l = Wr + (size_t)l * DDIM * DDIM;
        // projections
        gemm(h, Wq_l, nullptr, hp, NN, DDIM, DDIM, 0);
        gemm(rm, Wr_l, nullptr, rp, RRR, DDIM, DDIM, 0);
        // attention scalar scores
        proj_scores_kernel<<<(NN * HHEAD + TB - 1) / TB, TB, 0, stream>>>(
            hp, a_src + (size_t)l * DDIM, a_dst + (size_t)l * DDIM, ssrc, sdst, NN);
        proj_scores_kernel<<<(RRR * HHEAD + TB - 1) / TB, TB, 0, stream>>>(
            rp, a_rel + (size_t)l * DDIM, nullptr, srel, nullptr, RRR);
        // edge logits
        edge_logits_kernel<<<gEH, TB, 0, stream>>>(ssrc, sdst, srel, src, dst, rt, ebuf, sw);
        // iterative kth-max -> thr
        for (int it = 0; it < TOPK_C; ++it) {
            fill_u32_kernel<<<gNH, TB, 0, stream>>>(thr, ENC_NEG, nNH);
            seg_max_kernel<<<gEH, TB, 0, stream>>>(sw, dst, thr);
            if (it < TOPK_C - 1)
                suppress_kernel<<<gEH, TB, 0, stream>>>(sw, dst, thr);
        }
        // masked segment softmax
        fill_u32_kernel<<<gNH, TB, 0, stream>>>(mbits, ENC_NEG, nNH);
        seg_max_keep_kernel<<<gEH, TB, 0, stream>>>(ebuf, dst, thr, mbits);
        fill_u32_kernel<<<gNH, TB, 0, stream>>>((unsigned*)denom, 0u, nNH);
        exp_pass_kernel<<<gEH, TB, 0, stream>>>(ebuf, dst, thr, mbits, attn, denom);
        attn_norm_kernel<<<gEH, TB, 0, stream>>>(attn, dst, denom);
        // multi-hop diffusion
        hipMemcpyAsync(feat, hp, (size_t)NN * DDIM * sizeof(float),
                       hipMemcpyDeviceToDevice, stream);
        const int nND = NN * DDIM;
        const int gND = (nND + TB - 1) / TB;
        for (int hop = 0; hop < HOPS_C; ++hop) {
            fill_u32_kernel<<<gND, TB, 0, stream>>>((unsigned*)agg, 0u, nND);
            diffuse_scatter_kernel<<<(EE * 64 + TB - 1) / TB, TB, 0, stream>>>(
                feat, attn, src, dst, agg);
            axpby_kernel<<<gND, TB, 0, stream>>>(agg, hp, feat, 1.f - ALPHA_C, ALPHA_C, nND);
        }
        // residual + LN1
        residual_ln_kernel<<<NN, DDIM, 0, stream>>>(h, feat,
            ln1_g + (size_t)l * DDIM, ln1_b + (size_t)l * DDIM);
        // FFN (feat reused as tmp, agg reused as ff)
        gemm(h, ffn_w1 + (size_t)l * DDIM * DDIM, ffn_b1 + (size_t)l * DDIM, feat,
             NN, DDIM, DDIM, 1);
        gemm(feat, ffn_w2 + (size_t)l * DDIM * DDIM, ffn_b2 + (size_t)l * DDIM, agg,
             NN, DDIM, DDIM, 0);
        residual_ln_kernel<<<NN, DDIM, 0, stream>>>(h, agg,
            ln2_g + (size_t)l * DDIM, ln2_b + (size_t)l * DDIM);
    }

    // ---- classifier (X0 in feat, z1 in hp, z2 in agg) ----
    gather_pairs_kernel<<<(BBATCH * DDIM + TB - 1) / TB, TB, 0, stream>>>(h, xd, xp, feat);
    gemm(feat, cls_w1, cls_b1, hp, BBATCH, 256, 1024, 1);
    gemm(hp, cls_w2, cls_b2, agg, BBATCH, 1024, 512, 1);
    final_layer_kernel<<<(BBATCH * 32 + TB - 1) / TB, TB, 0, stream>>>(agg, cls_w3, cls_b3, out);
}
